// AutoregressiveRNN_68762426409192
// MI455X (gfx1250) — compile-verified, matrix operand order checked by reading, not run on hardware
//
#include <hip/hip_runtime.h>
#include <hip/hip_bf16.h>
#include <stdint.h>

// Problem constants (match reference file).
#define SEQ_LEN 8192
#define IDIM 256
#define HDIM 1024
#define NLAYER 4
#define SCAN_WGS 64   // persistent workgroups for recurrence kernels
#define HH ((size_t)HDIM * HDIM)

typedef __attribute__((ext_vector_type(16))) __bf16 v16bf;
typedef __attribute__((ext_vector_type(8)))  __bf16 v8bf;
typedef __attribute__((ext_vector_type(8)))  float  v8f;
typedef __attribute__((ext_vector_type(4)))  unsigned int v4u;
typedef __attribute__((ext_vector_type(8)))  int v8i;
typedef __attribute__((ext_vector_type(4)))  int v4i;

// ---------- small helpers ----------
__device__ __forceinline__ float bf2f(__bf16 v) {
  union { __bf16 b; unsigned short u; } c; c.b = v;
  return __uint_as_float((unsigned int)c.u << 16);
}
__device__ __forceinline__ __bf16 f2bf(float f) {
  unsigned int u = __float_as_uint(f);
  unsigned int r = (u + 0x7FFFu + ((u >> 16) & 1u)) >> 16;  // RNE
  union { unsigned short u; __bf16 b; } c; c.u = (unsigned short)r;
  return c.b;
}

// ---------- TDM: async tensor load Global -> LDS ----------
// Builds the D# descriptor (groups 0/1 per CDNA5 ISA 8.3/8.4) for a 2D tile
// of bf16 data: `rows` rows of `width` elements, row stride `stride` elems.
// One call == one TENSOR_LOAD_TO_LDS instruction (tracked by TENSORcnt).
// This toolchain exposes the 6-arg builtin: (g0 u32x4, g1 i32x8, g2 i32x4,
// g3 i32x4, i32x8, cpol). Groups 2/3 are NULL/zero for 2D tensors.
__device__ __forceinline__ void tdm_load_2d_bf16(const void* gptr,
                                                 unsigned int lds_off_bytes,
                                                 unsigned int width,
                                                 unsigned int rows,
                                                 unsigned int stride) {
  unsigned long long ga = (unsigned long long)(uintptr_t)gptr;
  v4u g0;
  g0[0] = 1u;                                   // count=1, user descriptor
  g0[1] = lds_off_bytes;                        // lds_addr
  g0[2] = (unsigned int)(ga & 0xFFFFFFFFu);     // global_addr[31:0]
  g0[3] = (unsigned int)((ga >> 32) & 0x1FFFFFFu) | (2u << 30);  // addr[56:32], type=2
  v8i g1;
  g1[0] = (int)(1u << 16);                      // wg_mask=0, data_size=1 (2B)
  g1[1] = (int)((width & 0xFFFFu) << 16);       // tensor_dim0[15:0]
  g1[2] = (int)((width >> 16) | ((rows & 0xFFFFu) << 16));   // td0[31:16], td1[15:0]
  g1[3] = (int)(((rows >> 16) & 0xFFFFu) | ((width & 0xFFFFu) << 16)); // td1 hi, tile_dim0
  g1[4] = (int)(rows & 0xFFFFu);                // tile_dim1=rows, tile_dim2=0
  g1[5] = (int)stride;                          // tensor_dim0_stride[31:0]
  g1[6] = 0;                                    // stride hi, dim1_stride lo
  g1[7] = 0;
  v4i z4 = {0, 0, 0, 0};
  v8i z8 = {0, 0, 0, 0, 0, 0, 0, 0};
  __builtin_amdgcn_tensor_load_to_lds(g0, g1, z4, z4, z8, 0);
}

// Low 32 bits of the generic address of a __shared__ object == LDS byte
// offset (flat addressing maps addr[31:0] directly into LDS).
__device__ __forceinline__ unsigned int lds_offset_of(const void* p) {
  return (unsigned int)(unsigned long long)(uintptr_t)p;
}

// Grid-wide barrier for a persistent kernel: monotonically increasing
// counter; release on arrive, acquire spin with s_sleep backoff.
__device__ __forceinline__ void grid_barrier(unsigned long long* ctr,
                                             unsigned long long expected) {
  __threadfence();
  __syncthreads();
  if (threadIdx.x == 0) {
    __hip_atomic_fetch_add(ctr, 1ull, __ATOMIC_RELEASE, __HIP_MEMORY_SCOPE_AGENT);
    while (__hip_atomic_load(ctr, __ATOMIC_ACQUIRE, __HIP_MEMORY_SCOPE_AGENT) < expected) {
      __builtin_amdgcn_s_sleep(2);
    }
  }
  __syncthreads();
}

// ---------- prep: fp32 -> bf16 conversions ----------
__global__ void convert_f32_bf16_kernel(const float* __restrict__ in,
                                        __bf16* __restrict__ out, size_t n) {
  size_t i = (size_t)blockIdx.x * blockDim.x + threadIdx.x;
  if (i < n) out[i] = f2bf(in[i]);
}

// in: K x N row-major fp32 -> out: N x K row-major bf16.
__global__ void transpose_f32_bf16_kernel(const float* __restrict__ in,
                                          __bf16* __restrict__ out,
                                          int K, int N) {
  size_t i = (size_t)blockIdx.x * blockDim.x + threadIdx.x;
  if (i >= (size_t)K * N) return;
  int k = (int)(i / N), n = (int)(i % N);
  out[(size_t)n * K + k] = f2bf(in[i]);
}

// ---------- WMMA GEMM: C[T,N] = A[T,K] @ Bt[N,K]^T + bias[N] ----------
// Each wave owns a 16x64 strip of C (4 N-tiles): the A fragment is loaded
// once per 32-k chunk and reused by 4 v_wmma_f32_16x16x32_bf16, cutting
// loads per WMMA from 4 to 2.5 and quadrupling wmma density in the loop.
__device__ __forceinline__ v16bf load_frag(const __bf16* p, int khalf) {
  v8bf lo = *(const v8bf*)(p + khalf);
  v8bf hi = *(const v8bf*)(p + 16 + khalf);
  v16bf r;
#pragma unroll
  for (int i = 0; i < 8; ++i) { r[i] = lo[i]; r[i + 8] = hi[i]; }
  return r;
}

__global__ __launch_bounds__(256) void wmma_gemm_bias_kernel(
    const __bf16* __restrict__ A, const __bf16* __restrict__ Bt,
    const float* __restrict__ bias, float* __restrict__ C, int K, int N) {
  const int lane   = threadIdx.x & 31;
  const int wave   = threadIdx.x >> 5;
  const int tileM  = blockIdx.x;
  const int tileN0 = (blockIdx.y * 8 + wave) * 4;   // 4 consecutive N-tiles
  if (tileN0 * 16 >= N) return;                     // idle waves (small N)
  const int mn    = lane & 15;
  const int khalf = (lane >> 4) << 3;
  const __bf16* arow = A + (size_t)(tileM * 16 + mn) * K;
  const __bf16* brow0 = Bt + (size_t)((tileN0 + 0) * 16 + mn) * K;
  const __bf16* brow1 = Bt + (size_t)((tileN0 + 1) * 16 + mn) * K;
  const __bf16* brow2 = Bt + (size_t)((tileN0 + 2) * 16 + mn) * K;
  const __bf16* brow3 = Bt + (size_t)((tileN0 + 3) * 16 + mn) * K;
  v8f acc[4] = {};
#pragma unroll 2
  for (int k = 0; k < K; k += 32) {
    v16bf a  = load_frag(arow + k, khalf);
    v16bf b0 = load_frag(brow0 + k, khalf);
    v16bf b1 = load_frag(brow1 + k, khalf);
    v16bf b2 = load_frag(brow2 + k, khalf);
    v16bf b3 = load_frag(brow3 + k, khalf);
    acc[0] = __builtin_amdgcn_wmma_f32_16x16x32_bf16(false, a, false, b0, (short)0, acc[0], false, false);
    acc[1] = __builtin_amdgcn_wmma_f32_16x16x32_bf16(false, a, false, b1, (short)0, acc[1], false, false);
    acc[2] = __builtin_amdgcn_wmma_f32_16x16x32_bf16(false, a, false, b2, (short)0, acc[2], false, false);
    acc[3] = __builtin_amdgcn_wmma_f32_16x16x32_bf16(false, a, false, b3, (short)0, acc[3], false, false);
  }
  const int rbase = (lane >> 4) << 3;  // C/D layout: M = r + 8*(lane>>4)
#pragma unroll
  for (int t = 0; t < 4; ++t) {
    const float bv = bias[(tileN0 + t) * 16 + mn];
#pragma unroll
    for (int r = 0; r < 8; ++r) {
      int row = tileM * 16 + rbase + r;
      C[(size_t)row * N + (tileN0 + t) * 16 + mn] = acc[t][r] + bv;
    }
  }
}

// ---------- open-loop recurrence scan (one layer) ----------
// 64 persistent WGs, 16 output columns each. The Wh column-slice (16x1024
// bf16 = 32KB) is DMA'd into LDS by the TDM once and stays resident for all
// 8192 steps; per step only the 4KB h vector moves through L2.
__global__ __launch_bounds__(256) void rnn_scan_kernel(
    const float* __restrict__ pre,    // SEQ x HDIM (bias already folded in)
    const __bf16* __restrict__ whT,   // HDIM x HDIM (N x K)
    __bf16* __restrict__ seq_bf,      // SEQ x HDIM  (layer output, bf16)
    float* __restrict__ h_last,       // HDIM        (final state, fp32)
    float* __restrict__ hbuf,         // 2 x HDIM ping-pong
    unsigned long long* __restrict__ ctr) {
  __shared__ __bf16 lds_wh[16 * HDIM];
  __shared__ float partial[256];
  const int tid = threadIdx.x;
  const int colbase = blockIdx.x * 16;
  const int nwg = gridDim.x;
  if (tid < 32) {  // wave 0 issues one TDM op for the whole 32KB slice
    tdm_load_2d_bf16(whT + (size_t)colbase * HDIM, lds_offset_of(lds_wh),
                     HDIM, 16, HDIM);
    __builtin_amdgcn_s_wait_tensorcnt(0);
  }
  if (tid < 16) hbuf[colbase + tid] = 0.f;  // h0 = 0
  __syncthreads();
  unsigned long long phase = 0;
  grid_barrier(ctr, (++phase) * nwg);
  const int c  = tid & 15;
  const int kb = (tid >> 4) * 64;
  for (int t = 0; t < SEQ_LEN; ++t) {
    const float* hp = hbuf + (t & 1) * HDIM;
    if (tid < 16 && t + 1 < SEQ_LEN)
      __builtin_prefetch(&pre[(size_t)(t + 1) * HDIM + colbase + tid], 0, 0);
    float s = 0.f;
    const __bf16* wrow = lds_wh + c * HDIM + kb;
#pragma unroll 8
    for (int j = 0; j < 64; ++j) s = fmaf(bf2f(wrow[j]), hp[kb + j], s);
    partial[tid] = s;
    __syncthreads();
    if (tid < 16) {
      float a = pre[(size_t)t * HDIM + colbase + tid];
#pragma unroll
      for (int g = 0; g < 16; ++g) a += partial[g * 16 + tid];
      float h = tanhf(a);
      hbuf[((t + 1) & 1) * HDIM + colbase + tid] = h;
      seq_bf[(size_t)t * HDIM + colbase + tid]   = f2bf(h);
      if (t == SEQ_LEN - 1) h_last[colbase + tid] = h;
    }
    grid_barrier(ctr, (++phase) * nwg);
  }
}

// ---------- autoregressive closed-loop generation ----------
// 64 persistent WGs. Every weight slice (4x Wh, 3x Wx, Wx0, out_W; ~240KB)
// is DMA'd into LDS by 9 TDM descriptors and stays resident, so the whole
// 2048-step loop does zero weight traffic. 5 grid barriers per step.
#define AR_LDS_ELEMS (4 * 16 * HDIM + 3 * 16 * HDIM + 16 * IDIM + 4 * HDIM)
#define AR_LDS_BYTES (AR_LDS_ELEMS * 2)

__global__ __launch_bounds__(256) void rnn_ar_kernel(
    const __bf16* __restrict__ whT_all,   // 4 x HDIM x HDIM (NxK each)
    const __bf16* __restrict__ wxT0,      // HDIM x IDIM
    const __bf16* __restrict__ wxT_rest,  // 3 x HDIM x HDIM
    const __bf16* __restrict__ outW_bf,   // IDIM x HDIM (already out,in)
    const float* __restrict__ b0, const float* __restrict__ b_rest,
    const float* __restrict__ out_b,
    float* __restrict__ h_state,          // 2 x 4 x HDIM (parity-buffered)
    float* __restrict__ x0_buf,           // 2 x IDIM
    const float* __restrict__ x_init,     // IDIM (= last open-loop output)
    float* __restrict__ out_ar,           // nsteps x IDIM
    unsigned long long* __restrict__ ctr,
    const int* __restrict__ n_steps_ptr) {
  extern __shared__ __bf16 smem[];
  __bf16* wh  = smem;                 // [4][16][HDIM]
  __bf16* wx  = wh + 4 * 16 * HDIM;   // [3][16][HDIM]
  __bf16* wx0 = wx + 3 * 16 * HDIM;   // [16][IDIM]
  __bf16* ow  = wx0 + 16 * IDIM;      // [4][HDIM]
  __shared__ float partial[256];
  const int tid = threadIdx.x;
  const int wg = blockIdx.x, nwg = gridDim.x;
  const int colbase = wg * 16;
  const int obase   = wg * 4;

  if (tid < 32) {  // wave 0 issues all weight-slice DMAs, then drains them
    for (int l = 0; l < 4; ++l)
      tdm_load_2d_bf16(whT_all + (size_t)l * HH + (size_t)colbase * HDIM,
                       lds_offset_of(wh + l * 16 * HDIM), HDIM, 16, HDIM);
    for (int l = 0; l < 3; ++l)
      tdm_load_2d_bf16(wxT_rest + (size_t)l * HH + (size_t)colbase * HDIM,
                       lds_offset_of(wx + l * 16 * HDIM), HDIM, 16, HDIM);
    tdm_load_2d_bf16(wxT0 + (size_t)colbase * IDIM, lds_offset_of(wx0),
                     IDIM, 16, IDIM);
    tdm_load_2d_bf16(outW_bf + (size_t)obase * HDIM, lds_offset_of(ow),
                     HDIM, 4, HDIM);
    __builtin_amdgcn_s_wait_tensorcnt(0);
  }
  if (wg == 0)
    for (int i = tid; i < IDIM; i += 256) x0_buf[i] = x_init[i];
  __syncthreads();

  unsigned long long phase = 0;
  grid_barrier(ctr, (++phase) * nwg);

  const int c = tid & 15, seg = tid >> 4;
  const int nsteps = n_steps_ptr[0];
  for (int s = 0; s < nsteps; ++s) {
    const int p = s & 1, q = p ^ 1;
    const float* hprev = h_state + (size_t)p * 4 * HDIM;
    float*       hnew  = h_state + (size_t)q * 4 * HDIM;
    // layer 0: h0' = tanh(x @ Wx0 + h0 @ Wh0 + b0)
    {
      const float* x = x0_buf + p * IDIM;
      const __bf16* wxr = wx0 + c * IDIM + seg * 16;
      const __bf16* whr = wh + c * HDIM + seg * 64;
      float acc = 0.f;
#pragma unroll 4
      for (int j = 0; j < 16; ++j) acc = fmaf(bf2f(wxr[j]), x[seg * 16 + j], acc);
#pragma unroll 4
      for (int j = 0; j < 64; ++j) acc = fmaf(bf2f(whr[j]), hprev[seg * 64 + j], acc);
      partial[tid] = acc; __syncthreads();
      if (tid < 16) {
        float a = b0[colbase + tid];
        for (int g = 0; g < 16; ++g) a += partial[g * 16 + tid];
        hnew[colbase + tid] = tanhf(a);
      }
      grid_barrier(ctr, (++phase) * nwg);
    }
    // layers 1..3
    for (int l = 1; l < 4; ++l) {
      const float* x = hnew + (l - 1) * HDIM;
      const float* h = hprev + l * HDIM;
      const __bf16* wxr = wx + ((l - 1) * 16 + c) * HDIM + seg * 64;
      const __bf16* whr = wh + (l * 16 + c) * HDIM + seg * 64;
      float acc = 0.f;
#pragma unroll 4
      for (int j = 0; j < 64; ++j) acc = fmaf(bf2f(wxr[j]), x[seg * 64 + j], acc);
#pragma unroll 4
      for (int j = 0; j < 64; ++j) acc = fmaf(bf2f(whr[j]), h[seg * 64 + j], acc);
      partial[tid] = acc; __syncthreads();
      if (tid < 16) {
        float a = b_rest[(size_t)(l - 1) * HDIM + colbase + tid];
        for (int g = 0; g < 16; ++g) a += partial[g * 16 + tid];
        hnew[l * HDIM + colbase + tid] = tanhf(a);
      }
      grid_barrier(ctr, (++phase) * nwg);
    }
    // out projection: x' = out_W @ h3' + out_b  (4 outputs per WG)
    {
      const float* h = hnew + 3 * HDIM;
      const int ol = tid & 3, s2 = tid >> 2;
      const __bf16* owr = ow + ol * HDIM + s2 * 16;
      float acc = 0.f;
#pragma unroll 4
      for (int j = 0; j < 16; ++j) acc = fmaf(bf2f(owr[j]), h[s2 * 16 + j], acc);
      partial[tid] = acc; __syncthreads();
      if (tid < 4) {
        float a = out_b[obase + tid];
        for (int g = 0; g < 64; ++g) a += partial[g * 4 + tid];
        x0_buf[q * IDIM + obase + tid] = a;
        out_ar[(size_t)s * IDIM + obase + tid] = a;
      }
      grid_barrier(ctr, (++phase) * nwg);
    }
  }
}

// ---------- host orchestration ----------
extern "C" void kernel_launch(void* const* d_in, const int* in_sizes, int n_in,
                              void* d_out, int out_size, void* d_ws, size_t ws_size,
                              hipStream_t stream) {
  const float* xs      = (const float*)d_in[0];
  const float* Wx0     = (const float*)d_in[1];
  const float* Wh0     = (const float*)d_in[2];
  const float* b0      = (const float*)d_in[3];
  const float* Wx_rest = (const float*)d_in[4];
  const float* Wh_rest = (const float*)d_in[5];
  const float* b_rest  = (const float*)d_in[6];
  const float* out_W   = (const float*)d_in[7];
  const float* out_b   = (const float*)d_in[8];
  const int*   n_steps = (const int*)d_in[9];
  float* out = (float*)d_out;

  char* ws = (char*)d_ws;
  size_t o = 0;
  auto alloc = [&](size_t bytes) -> size_t {
    size_t r = o; o = (o + bytes + 255) & ~(size_t)255; return r;
  };
  size_t off_ctr   = alloc(8 * sizeof(unsigned long long));
  size_t off_hbuf  = alloc(2 * HDIM * sizeof(float));
  size_t off_hst   = alloc(2 * NLAYER * HDIM * sizeof(float));
  size_t off_x0    = alloc(2 * IDIM * sizeof(float));
  size_t off_xsbf  = alloc((size_t)SEQ_LEN * IDIM * 2);
  size_t off_wxT0  = alloc((size_t)HDIM * IDIM * 2);
  size_t off_wxTr  = alloc(3 * HH * 2);
  size_t off_whT   = alloc(4 * HH * 2);
  size_t off_outW  = alloc((size_t)IDIM * HDIM * 2);
  size_t off_seqA  = alloc((size_t)SEQ_LEN * HDIM * 2);
  size_t off_seqB  = alloc((size_t)SEQ_LEN * HDIM * 2);
  size_t off_pre   = alloc((size_t)SEQ_LEN * HDIM * sizeof(float));
  if (ws_size < o) return;  // workspace too small

  unsigned long long* ctr = (unsigned long long*)(ws + off_ctr);
  float*  hbuf    = (float*)(ws + off_hbuf);
  float*  h_state = (float*)(ws + off_hst);
  float*  x0_buf  = (float*)(ws + off_x0);
  __bf16* xs_bf   = (__bf16*)(ws + off_xsbf);
  __bf16* wxT0    = (__bf16*)(ws + off_wxT0);
  __bf16* wxTr    = (__bf16*)(ws + off_wxTr);
  __bf16* whT     = (__bf16*)(ws + off_whT);
  __bf16* outWbf  = (__bf16*)(ws + off_outW);
  __bf16* seqA    = (__bf16*)(ws + off_seqA);
  __bf16* seqB    = (__bf16*)(ws + off_seqB);
  float*  pre     = (float*)(ws + off_pre);

  (void)hipMemsetAsync(ws + off_ctr, 0, 256, stream);

  // --- prep: bf16 conversions / transposes ---
  {
    size_t n = (size_t)SEQ_LEN * IDIM;
    convert_f32_bf16_kernel<<<(unsigned)((n + 255) / 256), 256, 0, stream>>>(xs, xs_bf, n);
  }
  transpose_f32_bf16_kernel<<<(unsigned)((IDIM * HDIM + 255) / 256), 256, 0, stream>>>(
      Wx0, wxT0, IDIM, HDIM);
  for (int l = 0; l < 3; ++l)
    transpose_f32_bf16_kernel<<<(unsigned)((HH + 255) / 256), 256, 0, stream>>>(
        Wx_rest + (size_t)l * HH, wxTr + (size_t)l * HH, HDIM, HDIM);
  transpose_f32_bf16_kernel<<<(unsigned)((HH + 255) / 256), 256, 0, stream>>>(
      Wh0, whT, HDIM, HDIM);
  for (int l = 0; l < 3; ++l)
    transpose_f32_bf16_kernel<<<(unsigned)((HH + 255) / 256), 256, 0, stream>>>(
        Wh_rest + (size_t)l * HH, whT + (size_t)(l + 1) * HH, HDIM, HDIM);
  {
    size_t n = (size_t)IDIM * HDIM;
    convert_f32_bf16_kernel<<<(unsigned)((n + 255) / 256), 256, 0, stream>>>(out_W, outWbf, n);
  }

  // --- open loop: GEMM (WMMA) + sequential scan, per layer ---
  dim3 gH(SEQ_LEN / 16, ((HDIM / 64) + 7) / 8);  // 512 x 2 (4 N-tiles/wave)
  wmma_gemm_bias_kernel<<<gH, 256, 0, stream>>>(xs_bf, wxT0, b0, pre, IDIM, HDIM);
  rnn_scan_kernel<<<SCAN_WGS, 256, 0, stream>>>(pre, whT, seqA, h_state, hbuf, ctr + 0);

  const __bf16* seq_in = seqA;
  __bf16*       seq_out = seqB;
  for (int l = 1; l < NLAYER; ++l) {
    wmma_gemm_bias_kernel<<<gH, 256, 0, stream>>>(
        seq_in, wxTr + (size_t)(l - 1) * HH, b_rest + (size_t)(l - 1) * HDIM,
        pre, HDIM, HDIM);
    rnn_scan_kernel<<<SCAN_WGS, 256, 0, stream>>>(
        pre, whT + (size_t)l * HH, seq_out, h_state + (size_t)l * HDIM, hbuf, ctr + l);
    const __bf16* t = seq_in; seq_in = seq_out; seq_out = (__bf16*)t;
  }

  // --- output projection for the whole open-loop sequence ---
  dim3 gO(SEQ_LEN / 16, ((IDIM / 64) + 7) / 8);  // 512 x 1
  wmma_gemm_bias_kernel<<<gO, 256, 0, stream>>>(seq_in, outWbf, out_b, out, HDIM, IDIM);

  // --- autoregressive generation (persistent, all weights LDS-resident) ---
  (void)hipFuncSetAttribute((const void*)rnn_ar_kernel,
                            hipFuncAttributeMaxDynamicSharedMemorySize, AR_LDS_BYTES);
  rnn_ar_kernel<<<SCAN_WGS, 256, AR_LDS_BYTES, stream>>>(
      whT, wxT0, wxTr, outWbf, b0, b_rest, out_b,
      h_state, x0_buf,
      out + (size_t)(SEQ_LEN - 1) * IDIM,   // x_init = last open-loop output
      out + (size_t)SEQ_LEN * IDIM,         // AR outputs appended
      ctr + 4, n_steps);
}
